// SelfAttention_13829794693416
// MI455X (gfx1250) — matrix operand order repaired and checked by reading, not compiled, not run on hardware
//
#include <hip/hip_runtime.h>
#include <cmath>

// ---------------------------------------------------------------------------
// CDNA5 (gfx1250) self-attention (SAGAN block), flash-attention style, bf16 WMMA
// + GLOBAL_LOAD_ASYNC_TO_LDS tile staging, ds_swizzle butterfly reductions
// ---------------------------------------------------------------------------

typedef __attribute__((ext_vector_type(16))) __bf16 v16bf;
typedef __attribute__((ext_vector_type(8)))  float  v8f;

#define TPB 256           // 8 waves of 32
#define B_   16
#define HH   64
#define WW   64
#define C_   512
#define N_   4096         // H*W
#define ND_  1024         // pooled
#define DK_  64           // C/8
#define DV_  256          // C/2

static __device__ __forceinline__ v8f wmma_bf16(v16bf a, v16bf b, v8f c) {
    // emits v_wmma_f32_16x16x32_bf16
    return __builtin_amdgcn_wmma_f32_16x16x32_bf16(
        /*neg_a=*/false, a, /*neg_b=*/false, b,
        /*c_mod=*/(short)0, c, /*reuse_a=*/false, /*reuse_b=*/false);
}

// 16-bit A-matrix 16x32 layout (ISA 7.12.2): element e (0..15) of lane-half h
static __device__ __forceinline__ int a_kidx(int e, int h) {
    return ((e < 8) ? 0 : 16) + h * 8 + (((e >> 1) & 3) << 1) + (e & 1);
}
// 16-bit B-matrix 32x16 layout: same k order as A (element e of lane-half h)
static __device__ __forceinline__ int b_kidx(int e, int h) { return ((e < 8) ? 0 : 16) + h * 8 + (e & 7); }

// ---- ds_swizzle xor-butterfly (group-of-32 mode: imm = xor<<10 | and 0x1f) ----
template <int IMM>
static __device__ __forceinline__ float swz(float v) {
    return __int_as_float(__builtin_amdgcn_ds_swizzle(__float_as_int(v), IMM));
}
static __device__ __forceinline__ float red_max16(float v) {
    v = fmaxf(v, swz<0x041F>(v));   // xor 1
    v = fmaxf(v, swz<0x081F>(v));   // xor 2
    v = fmaxf(v, swz<0x101F>(v));   // xor 4
    v = fmaxf(v, swz<0x201F>(v));   // xor 8  (stays within 16-lane half)
    return v;
}
static __device__ __forceinline__ float red_sum16(float v) {
    v += swz<0x041F>(v);
    v += swz<0x081F>(v);
    v += swz<0x101F>(v);
    v += swz<0x201F>(v);
    return v;
}

// ---- CDNA5 async global->LDS DMA (ASYNCcnt) ----
static __device__ __forceinline__ unsigned lds_off(const void* p) {
    // LDS aperture base has zero low 32 bits -> low bits of flat addr == LDS offset
    return (unsigned)(uintptr_t)p;
}
static __device__ __forceinline__ void async_copy16(unsigned ldsAddr, const void* gptr) {
    asm volatile("global_load_async_to_lds_b128 %0, %1, off"
                 :: "v"(ldsAddr), "v"(gptr) : "memory");
}
static __device__ __forceinline__ void async_wait0() {
    asm volatile("s_wait_asynccnt 0" ::: "memory");
}

// ---------------------------------------------------------------------------
// Spectral norm scale (one block per weight): scale = 1/sigma
// ---------------------------------------------------------------------------
__device__ float block_reduce_sum(float val, float* red) {
    int tid = threadIdx.x;
    red[tid] = val;
    __syncthreads();
    for (int s = TPB / 2; s > 0; s >>= 1) {
        if (tid < s) red[tid] += red[tid + s];
        __syncthreads();
    }
    float r = red[0];
    __syncthreads();
    return r;
}

__global__ void __launch_bounds__(TPB)
sn_scale_kernel(const float* __restrict__ W, const float* __restrict__ u,
                int C, int F, float* __restrict__ scale_out) {
    __shared__ float v[512];
    __shared__ float t[512];
    __shared__ float red[TPB];
    const int tid = threadIdx.x;
    for (int c = tid; c < C; c += TPB) {
        float s = 0.f;
        for (int f = 0; f < F; ++f) s += u[f] * W[(size_t)c * F + f];
        v[c] = s;
    }
    __syncthreads();
    float loc = 0.f;
    for (int c = tid; c < C; c += TPB) loc += v[c] * v[c];
    float nv = sqrtf(block_reduce_sum(loc, red));
    for (int c = tid; c < C; c += TPB) v[c] /= (nv + 1e-12f);
    __syncthreads();
    for (int f = tid; f < F; f += TPB) {
        float s = 0.f;
        for (int c = 0; c < C; ++c) s += v[c] * W[(size_t)c * F + f];
        t[f] = s;
    }
    __syncthreads();
    loc = 0.f;
    for (int f = tid; f < F; f += TPB) loc += t[f] * t[f];
    float nt = sqrtf(block_reduce_sum(loc, red));
    if (tid == 0) {
        float sigma = nt * nt / (nt + 1e-12f);
        scale_out[0] = 1.0f / sigma;
    }
}

__global__ void __launch_bounds__(TPB)
wscale_bf16_kernel(const float* __restrict__ W, const float* __restrict__ scale,
                   __bf16* __restrict__ out, int n) {
    float s = scale[0];
    for (int i = blockIdx.x * blockDim.x + threadIdx.x; i < n;
         i += gridDim.x * blockDim.x)
        out[i] = (__bf16)(W[i] * s);
}

// ---------------------------------------------------------------------------
// Projection GEMM: Y[65536, COUT] = bf16(X[65536,512]) @ Wbf[512,COUT]
// ---------------------------------------------------------------------------
template <int COUT>
__global__ void __launch_bounds__(TPB, 1)
proj_kernel(const float* __restrict__ x, const __bf16* __restrict__ wbf,
            __bf16* __restrict__ y) {
    constexpr int NT = COUT / 32;
    __shared__ __bf16 ldsA[64][32];
    __shared__ __bf16 ldsB[32][COUT];

    const int tid   = threadIdx.x;
    const int wid   = tid >> 5;
    const int lane  = tid & 31;
    const int m_idx = wid & 3;
    const int n0    = (wid >> 2) * (COUT / 2);
    const int h     = lane >> 4;
    const int ln    = lane & 15;
    const size_t mBase = (size_t)blockIdx.x * 64;

    v8f acc[NT];
#pragma unroll
    for (int t = 0; t < NT; ++t) acc[t] = (v8f){};

    for (int kb = 0; kb < C_; kb += 32) {
        // stage W tile (32 x COUT bf16) via async DMA to LDS
        constexpr int SEGR = COUT / 8;              // 16B segments per row
#pragma unroll
        for (int i = 0; i < (32 * SEGR) / TPB; ++i) {
            int idx = tid + i * TPB;
            int r = idx / SEGR, seg = idx % SEGR;
            async_copy16(lds_off(&ldsB[r][0]) + seg * 16,
                         (const char*)(wbf + (size_t)(kb + r) * COUT) + seg * 16);
        }
        // stage X tile (64x32) as bf16 (needs f32->bf16 convert, via VGPRs)
#pragma unroll
        for (int i = 0; i < (64 * 32) / TPB; ++i) {
            int idx = tid + i * TPB;
            int r = idx >> 5, c = idx & 31;
            ldsA[r][c] = (__bf16)x[(mBase + r) * C_ + kb + c];
        }
        async_wait0();
        __syncthreads();

        v16bf a;
#pragma unroll
        for (int e = 0; e < 16; ++e) a[e] = ldsA[m_idx * 16 + ln][a_kidx(e, h)];
#pragma unroll
        for (int t = 0; t < NT; ++t) {
            v16bf b;
#pragma unroll
            for (int e = 0; e < 16; ++e) b[e] = ldsB[b_kidx(e, h)][n0 + t * 16 + ln];
            acc[t] = wmma_bf16(a, b, acc[t]);
        }
        __syncthreads();
    }
#pragma unroll
    for (int t = 0; t < NT; ++t)
#pragma unroll
        for (int r = 0; r < 8; ++r) {
            int m = m_idx * 16 + r + h * 8;
            int n = n0 + t * 16 + ln;
            y[(mBase + m) * COUT + n] = (__bf16)acc[t][r];
        }
}

// ---------------------------------------------------------------------------
// 2x2 max pool on bf16 NHWC
// ---------------------------------------------------------------------------
__global__ void __launch_bounds__(TPB)
pool2x2_kernel(const __bf16* __restrict__ in, __bf16* __restrict__ out,
               int C, int total) {
    for (int idx = blockIdx.x * blockDim.x + threadIdx.x; idx < total;
         idx += gridDim.x * blockDim.x) {
        int c  = idx % C;
        int wp = (idx / C) % 32;
        int hp = (idx / (C * 32)) % 32;
        int b  = idx / (C * 1024);
        float m = -1e30f;
#pragma unroll
        for (int dy = 0; dy < 2; ++dy)
#pragma unroll
            for (int dx = 0; dx < 2; ++dx) {
                size_t n = ((size_t)(b * HH + hp * 2 + dy)) * WW + (wp * 2 + dx);
                m = fmaxf(m, (float)in[n * C + c]);
            }
        out[idx] = (__bf16)m;
    }
}

// ---------------------------------------------------------------------------
// Fused flash attention + final conv + residual. Block owns 64 query rows.
// waves: m_idx = wid&3 (16 rows), v_idx = wid>>2 (128 value cols)
// ---------------------------------------------------------------------------
__global__ void __launch_bounds__(TPB, 1)
attn_kernel(const __bf16* __restrict__ theta, const __bf16* __restrict__ phi,
            const __bf16* __restrict__ g, const __bf16* __restrict__ wattn,
            const float* __restrict__ x, const float* __restrict__ sigma_ptr,
            float* __restrict__ out) {
    __shared__ __bf16 thLds[64][DK_];        // 8 KB
    __shared__ __bf16 phiLds[64][DK_];       // 8 KB
    __shared__ __bf16 pLds[8][16][64];       // 16 KB, per-wave P scratch
    __shared__ __bf16 gLds[64][DV_];         // 32 KB (reused for AG in epilogue)

    const int tid   = threadIdx.x;
    const int wid   = tid >> 5;
    const int lane  = tid & 31;
    const int m_idx = wid & 3;
    const int v_idx = wid >> 2;
    const int h     = lane >> 4;
    const int ln    = lane & 15;
    const int b     = blockIdx.x >> 6;
    const int mBase = (blockIdx.x & 63) * 64;

    // ---- stage theta tile (64x64 bf16 = 8KB) via async DMA ----
#pragma unroll
    for (int i = 0; i < 2; ++i) {
        int idx = tid + i * TPB;
        int r = idx >> 3, seg = idx & 7;
        async_copy16(lds_off(&thLds[r][0]) + seg * 16,
                     (const char*)(theta + ((size_t)b * N_ + mBase + r) * DK_) + seg * 16);
    }
    async_wait0();
    __syncthreads();
    v16bf ath[2];
#pragma unroll
    for (int f = 0; f < 2; ++f)
#pragma unroll
        for (int e = 0; e < 16; ++e)
            ath[f][e] = thLds[m_idx * 16 + ln][f * 32 + a_kidx(e, h)];

    // ---- flash state ----
    float rm[8], rl[8];
#pragma unroll
    for (int r = 0; r < 8; ++r) { rm[r] = -1e30f; rl[r] = 0.f; }
    v8f acc[8];
#pragma unroll
    for (int t = 0; t < 8; ++t) acc[t] = (v8f){};

    for (int nd0 = 0; nd0 < ND_; nd0 += 64) {
        __syncthreads();
        // stage phi tile (64x64) and g tile (64x256) via async DMA
#pragma unroll
        for (int i = 0; i < 2; ++i) {
            int idx = tid + i * TPB;
            int r = idx >> 3, seg = idx & 7;
            async_copy16(lds_off(&phiLds[r][0]) + seg * 16,
                         (const char*)(phi + ((size_t)b * ND_ + nd0 + r) * DK_) + seg * 16);
        }
#pragma unroll
        for (int i = 0; i < 8; ++i) {
            int idx = tid + i * TPB;
            int r = idx >> 5, seg = idx & 31;
            async_copy16(lds_off(&gLds[r][0]) + seg * 16,
                         (const char*)(g + ((size_t)b * ND_ + nd0 + r) * DV_) + seg * 16);
        }
        async_wait0();
        __syncthreads();

        // ---- S = theta (16x64) x phi^T : 4 N-tiles, K=64 ----
        v8f s[4];
#pragma unroll
        for (int t = 0; t < 4; ++t) s[t] = (v8f){};
#pragma unroll
        for (int f = 0; f < 2; ++f)
#pragma unroll
            for (int t = 0; t < 4; ++t) {
                v16bf bb;  // B(k,n) = phi(nd_local = n, k)
#pragma unroll
                for (int e = 0; e < 16; ++e)
                    bb[e] = phiLds[t * 16 + ln][f * 32 + b_kidx(e, h)];
                s[t] = wmma_bf16(ath[f], bb, s[t]);
            }

        // ---- online softmax over this 64-wide slab ----
#pragma unroll
        for (int r = 0; r < 8; ++r) {
            float tm = fmaxf(fmaxf(s[0][r], s[1][r]), fmaxf(s[2][r], s[3][r]));
            tm = red_max16(tm);
            float nm = fmaxf(rm[r], tm);
            float alpha = __expf(rm[r] - nm);
            float psum = 0.f;
#pragma unroll
            for (int t = 0; t < 4; ++t) {
                float p = __expf(s[t][r] - nm);
                s[t][r] = p;
                psum += p;
            }
            psum = red_sum16(psum);
            rl[r] = rl[r] * alpha + psum;
            rm[r] = nm;
#pragma unroll
            for (int vt = 0; vt < 8; ++vt) acc[vt][r] *= alpha;
        }

        // ---- P (16x64, D layout) -> per-wave LDS -> A layout fragments ----
#pragma unroll
        for (int t = 0; t < 4; ++t)
#pragma unroll
            for (int r = 0; r < 8; ++r)
                pLds[wid][r + h * 8][t * 16 + ln] = (__bf16)s[t][r];

        // ---- acc += P (16x64) @ G (64x128 half) ----
#pragma unroll
        for (int f = 0; f < 2; ++f) {
            v16bf pa;
#pragma unroll
            for (int e = 0; e < 16; ++e)
                pa[e] = pLds[wid][ln][f * 32 + a_kidx(e, h)];
#pragma unroll
            for (int vt = 0; vt < 8; ++vt) {
                v16bf gb;
                int col = v_idx * 128 + vt * 16 + ln;
#pragma unroll
                for (int e = 0; e < 16; ++e)
                    gb[e] = gLds[f * 32 + b_kidx(e, h)][col];
                acc[vt] = wmma_bf16(pa, gb, acc[vt]);
            }
        }
    }

    // ---- normalize and stash AG (64x256) into gLds ----
    __syncthreads();
#pragma unroll
    for (int vt = 0; vt < 8; ++vt)
#pragma unroll
        for (int r = 0; r < 8; ++r) {
            int row = m_idx * 16 + r + h * 8;
            int col = v_idx * 128 + vt * 16 + ln;
            gLds[row][col] = (__bf16)(acc[vt][r] / rl[r]);
        }
    __syncthreads();

    // ---- fused final conv: OUT(64x512) = AG(64x256) @ Wattn(256x512), + x + sigma ----
    const float sig = sigma_ptr[0];
    const int m2 = (wid & 3) * 16;
    const int c0 = (wid >> 2) * 256;
    for (int nt0 = 0; nt0 < 4; ++nt0) {
        v8f acc2[4];
#pragma unroll
        for (int t = 0; t < 4; ++t) acc2[t] = (v8f){};
        for (int kb = 0; kb < DV_; kb += 32) {
            v16bf aa;
#pragma unroll
            for (int e = 0; e < 16; ++e)
                aa[e] = gLds[m2 + ln][kb + a_kidx(e, h)];
#pragma unroll
            for (int t = 0; t < 4; ++t) {
                v16bf bb;
                int col = c0 + nt0 * 64 + t * 16 + ln;
#pragma unroll
                for (int e = 0; e < 16; ++e)
                    bb[e] = wattn[(size_t)(kb + b_kidx(e, h)) * C_ + col];
                acc2[t] = wmma_bf16(aa, bb, acc2[t]);
            }
        }
#pragma unroll
        for (int t = 0; t < 4; ++t)
#pragma unroll
            for (int r = 0; r < 8; ++r) {
                int row = mBase + m2 + r + h * 8;
                int col = c0 + nt0 * 64 + t * 16 + ln;
                size_t o = ((size_t)b * N_ + row) * C_ + col;
                out[o] = x[o] + sig * acc2[t][r];
            }
    }
}

// ---------------------------------------------------------------------------
// Launch
// ---------------------------------------------------------------------------
extern "C" void kernel_launch(void* const* d_in, const int* in_sizes, int n_in,
                              void* d_out, int out_size, void* d_ws, size_t ws_size,
                              hipStream_t stream) {
    (void)in_sizes; (void)n_in; (void)out_size; (void)ws_size;
    const float* x       = (const float*)d_in[0];
    const float* w_theta = (const float*)d_in[1];
    const float* u_theta = (const float*)d_in[2];
    const float* w_phi   = (const float*)d_in[3];
    const float* u_phi   = (const float*)d_in[4];
    const float* w_g     = (const float*)d_in[5];
    const float* u_g     = (const float*)d_in[6];
    const float* w_attn  = (const float*)d_in[7];
    const float* u_attn  = (const float*)d_in[8];
    const float* sigma   = (const float*)d_in[9];
    float* out = (float*)d_out;

    char* ws = (char*)d_ws;
    size_t off = 0;
    float* scales = (float*)(ws + off);          off += 256;
    __bf16* wbf_th = (__bf16*)(ws + off);        off += (size_t)C_ * DK_ * 2;
    __bf16* wbf_ph = (__bf16*)(ws + off);        off += (size_t)C_ * DK_ * 2;
    __bf16* wbf_g  = (__bf16*)(ws + off);        off += (size_t)C_ * DV_ * 2;
    __bf16* wbf_at = (__bf16*)(ws + off);        off += (size_t)DV_ * C_ * 2;
    __bf16* theta  = (__bf16*)(ws + off);        off += (size_t)B_ * N_ * DK_ * 2;
    __bf16* phi_f  = (__bf16*)(ws + off);        off += (size_t)B_ * N_ * DK_ * 2;
    __bf16* g_f    = (__bf16*)(ws + off);        off += (size_t)B_ * N_ * DV_ * 2;
    __bf16* phi_p  = (__bf16*)(ws + off);        off += (size_t)B_ * ND_ * DK_ * 2;
    __bf16* g_p    = (__bf16*)(ws + off);        off += (size_t)B_ * ND_ * DV_ * 2;

    // 1) spectral norm scales
    sn_scale_kernel<<<1, TPB, 0, stream>>>(w_theta, u_theta, C_, DK_, scales + 0);
    sn_scale_kernel<<<1, TPB, 0, stream>>>(w_phi,   u_phi,   C_, DK_, scales + 1);
    sn_scale_kernel<<<1, TPB, 0, stream>>>(w_g,     u_g,     C_, DV_, scales + 2);
    sn_scale_kernel<<<1, TPB, 0, stream>>>(w_attn,  u_attn,  DV_, C_, scales + 3);

    // 2) scaled bf16 weight copies
    wscale_bf16_kernel<<<(C_ * DK_ + TPB - 1) / TPB, TPB, 0, stream>>>(w_theta, scales + 0, wbf_th, C_ * DK_);
    wscale_bf16_kernel<<<(C_ * DK_ + TPB - 1) / TPB, TPB, 0, stream>>>(w_phi,   scales + 1, wbf_ph, C_ * DK_);
    wscale_bf16_kernel<<<(C_ * DV_ + TPB - 1) / TPB, TPB, 0, stream>>>(w_g,     scales + 2, wbf_g,  C_ * DV_);
    wscale_bf16_kernel<<<(DV_ * C_ + TPB - 1) / TPB, TPB, 0, stream>>>(w_attn,  scales + 3, wbf_at, DV_ * C_);

    // 3) projections (1x1 convs) via bf16 WMMA
    const int mBlocks = (B_ * N_) / 64;  // 1024
    proj_kernel<DK_><<<mBlocks, TPB, 0, stream>>>(x, wbf_th, theta);
    proj_kernel<DK_><<<mBlocks, TPB, 0, stream>>>(x, wbf_ph, phi_f);
    proj_kernel<DV_><<<mBlocks, TPB, 0, stream>>>(x, wbf_g,  g_f);

    // 4) 2x2 max pools
    {
        int tot = B_ * ND_ * DK_;
        pool2x2_kernel<<<(tot + TPB - 1) / TPB, TPB, 0, stream>>>(phi_f, phi_p, DK_, tot);
    }
    {
        int tot = B_ * ND_ * DV_;
        pool2x2_kernel<<<(tot + TPB - 1) / TPB, TPB, 0, stream>>>(g_f, g_p, DV_, tot);
    }

    // 5) fused flash attention + final conv + residual
    attn_kernel<<<B_ * (N_ / 64), TPB, 0, stream>>>(theta, phi_p, g_p, wbf_at, x, sigma, out);
}
